// EGNNDynamics_6493990552277
// MI455X (gfx1250) — compile-verified, matrix operand order checked by reading, not run on hardware
//
#include <hip/hip_runtime.h>
#include <hip/hip_bf16.h>
#include <hip/hip_fp16.h>

typedef __attribute__((ext_vector_type(16))) _Float16 v16h;
typedef __attribute__((ext_vector_type(8)))  float    v8f;

#define HIDDEN   64
#define JOINTF   16
#define NLIG     512
#define NPOC     1024
#define ATOM_NF  16
#define RES_NF   32
#define NORM_F   100.0f
#define NORM_CC  1.0f
#define JITTER_V 1e-4f

// fast reciprocal: v_rcp_f32 (1 ulp) instead of IEEE div ladder
__device__ __forceinline__ float frcp_f(float x) { return __builtin_amdgcn_rcpf(x); }
__device__ __forceinline__ float silu_f(float x) { return x * frcp_f(1.0f + __expf(-x)); }
__device__ __forceinline__ float ntn_f(float v) {
    if (isnan(v)) return 0.0f;
    if (isinf(v)) return v > 0.0f ? 1.0f : -1.0f;
    return v;
}

// ---------------- small VALU kernels ----------------

__global__ void zero_kernel(float* p, int n) {
    int t = blockIdx.x * blockDim.x + threadIdx.x;
    if (t < n) p[t] = 0.0f;
}

// ligand encoder: h_lt[i] = [LN(mlp2(h_l, enc_a)), t[mask_lig[i]]]  (17 feats)
__global__ void enc_lig_kernel(const float* xh_lig, const float* tvec, const int* mask_lig,
                               const float* w1, const float* b1, const float* w2, const float* b2,
                               float* h_lt) {
    int i = blockIdx.x * blockDim.x + threadIdx.x;
    if (i >= NLIG) return;
    float h[ATOM_NF];
    for (int f = 0; f < ATOM_NF; ++f) h[f] = xh_lig[i * 19 + 3 + f];
    float t1[2 * ATOM_NF];
    for (int k = 0; k < 2 * ATOM_NF; ++k) {
        float a = b1[k];
        for (int f = 0; f < ATOM_NF; ++f) a += h[f] * w1[f * (2 * ATOM_NF) + k];
        t1[k] = silu_f(a);
    }
    float e2[JOINTF]; float mean = 0.0f;
    for (int k = 0; k < JOINTF; ++k) {
        float a = b2[k];
        for (int f = 0; f < 2 * ATOM_NF; ++f) a += t1[f] * w2[f * JOINTF + k];
        e2[k] = a; mean += a;
    }
    mean *= (1.0f / JOINTF);
    float var = 0.0f;
    for (int k = 0; k < JOINTF; ++k) { float d = e2[k] - mean; var += d * d; }
    var *= (1.0f / JOINTF);
    float inv = rsqrtf(var + 1e-5f);
    for (int k = 0; k < JOINTF; ++k) h_lt[i * 17 + k] = (e2[k] - mean) * inv;
    h_lt[i * 17 + 16] = tvec[mask_lig[i]];
}

// pocket encoder: h_p_emb = LN(mlp2(h_p, enc_p)); hp_proj = h_p_emb @ wp + bp; x_p extract
__global__ void enc_poc_kernel(const float* xh_ctx,
                               const float* w1, const float* b1, const float* w2, const float* b2,
                               const float* wp, const float* bp,
                               float* h_p_emb, float* hp_proj, float* x_p) {
    int j = blockIdx.x * blockDim.x + threadIdx.x;
    if (j >= NPOC) return;
    const int IN = 3 + RES_NF;
    float h[RES_NF];
    for (int f = 0; f < RES_NF; ++f) h[f] = xh_ctx[j * IN + 3 + f];
    for (int d = 0; d < 3; ++d) x_p[j * 3 + d] = xh_ctx[j * IN + d];
    float t1[2 * RES_NF];
    for (int k = 0; k < 2 * RES_NF; ++k) {
        float a = b1[k];
        for (int f = 0; f < RES_NF; ++f) a += h[f] * w1[f * (2 * RES_NF) + k];
        t1[k] = silu_f(a);
    }
    float e2[JOINTF]; float mean = 0.0f;
    for (int k = 0; k < JOINTF; ++k) {
        float a = b2[k];
        for (int f = 0; f < 2 * RES_NF; ++f) a += t1[f] * w2[f * JOINTF + k];
        e2[k] = a; mean += a;
    }
    mean *= (1.0f / JOINTF);
    float var = 0.0f;
    for (int k = 0; k < JOINTF; ++k) { float d = e2[k] - mean; var += d * d; }
    var *= (1.0f / JOINTF);
    float inv = rsqrtf(var + 1e-5f);
    float e[JOINTF];
    for (int k = 0; k < JOINTF; ++k) { e[k] = (e2[k] - mean) * inv; h_p_emb[j * JOINTF + k] = e[k]; }
    for (int k = 0; k < HIDDEN; ++k) {
        float a = bp[k];
        for (int f = 0; f < JOINTF; ++f) a += e[f] * wp[f * HIDDEN + k];
        hp_proj[j * HIDDEN + k] = a;
    }
}

// x_l = x + jitter*noise; h_ll = h_lt@wi+bi; h_cr = h_lt@wl+bl
__global__ void lig_init_kernel(const float* xh_lig, const float* noise, const float* h_lt,
                                const float* wi, const float* bi, const float* wl, const float* bl,
                                float* x_l0, float* x_ll, float* x_cr, float* h_ll, float* h_cr) {
    int i = blockIdx.x * blockDim.x + threadIdx.x;
    if (i >= NLIG) return;
    for (int d = 0; d < 3; ++d) {
        float x = xh_lig[i * 19 + d] + JITTER_V * noise[i * 3 + d];
        x_l0[i * 3 + d] = x; x_ll[i * 3 + d] = x; x_cr[i * 3 + d] = x;
    }
    float hl[17];
    for (int f = 0; f < 17; ++f) hl[f] = h_lt[i * 17 + f];
    for (int k = 0; k < HIDDEN; ++k) {
        float a = bi[k], b = bl[k];
        for (int f = 0; f < 17; ++f) { a += hl[f] * wi[f * HIDDEN + k]; b += hl[f] * wl[f * HIDDEN + k]; }
        h_ll[i * HIDDEN + k] = a; h_cr[i * HIDDEN + k] = b;
    }
}

// out[i,k] = sum_f h[i,f] * w[f,k]   (64x64 projection, no bias)
__global__ void proj_kernel(int N, const float* h, const float* w, float* out) {
    int t = blockIdx.x * blockDim.x + threadIdx.x;
    if (t >= N * HIDDEN) return;
    int i = t >> 6, k = t & 63;
    const float* hr = h + i * HIDDEN;
    float a = 0.0f;
    for (int f = 0; f < HIDDEN; ++f) a += hr[f] * w[f * HIDDEN + k];
    out[t] = a;
}

// ---------------- fused EGCL edge kernel (WMMA) ----------------
// Block: 128 threads (4 wave32); owns 16 receivers; loops senders in tiles of 8.
// Edge tile = 128 edges; two [128,64]@[64,64] matmuls via v_wmma_f32_16x16x32_f16.
__global__ __launch_bounds__(128) void egcl_edge_kernel(
        int Nj,
        const float* hi_proj, const float* hj_proj,
        const float* xi, const float* xj,
        const int* mi, const int* mj, float cutoff,
        const float* ew1, const float* eb1,
        const float* ew2, const float* eb2,
        const float* cw1, const float* cb1, const float* cw2,
        float* agg, float* trans) {
    __shared__ _Float16 sA[128][HIDDEN];   // edge features e
    __shared__ _Float16 sM[128][HIDDEN];   // messages m
    __shared__ _Float16 sW2[HIDDEN][HIDDEN];
    __shared__ _Float16 sC1[HIDDEN][HIDDEN];
    __shared__ _Float16 sHi[16][HIDDEN];
    __shared__ _Float16 sHj[8][HIDDEN];
    __shared__ float sDiff[128][3];
    __shared__ float sAdj[128];
    __shared__ float sD2[128];
    __shared__ float sPhi[128];
    __shared__ float sAgg[16][HIDDEN];
    __shared__ float sTrn[16][3];
    __shared__ float sEb1[HIDDEN], sWd2[HIDDEN], sEb2[HIDDEN], sCb1[HIDDEN], sCw2[HIDDEN];

    const int tid  = threadIdx.x;
    const int lane = tid & 31;
    const int wid  = tid >> 5;
    const int nb   = wid * 16;           // this wave's N-tile
    const int half = lane >> 4;
    const int i0   = blockIdx.x * 16;
    const float* wd2 = ew1 + 128 * HIDDEN;   // ew1 row 2*HID (d2 weight)

    for (int idx = tid; idx < HIDDEN * HIDDEN; idx += 128) {
        sW2[idx >> 6][idx & 63] = (_Float16)ew2[idx];
        sC1[idx >> 6][idx & 63] = (_Float16)cw1[idx];
    }
    if (tid < HIDDEN) {
        sEb1[tid] = eb1[tid]; sWd2[tid] = wd2[tid]; sEb2[tid] = eb2[tid];
        sCb1[tid] = cb1[tid]; sCw2[tid] = cw2[tid];
    }
    for (int idx = tid; idx < 16 * HIDDEN; idx += 128) {
        sHi[idx >> 6][idx & 63] = (_Float16)hi_proj[(i0 + (idx >> 6)) * HIDDEN + (idx & 63)];
        sAgg[idx >> 6][idx & 63] = 0.0f;
    }
    if (tid < 48) sTrn[tid / 3][tid % 3] = 0.0f;
    __syncthreads();

    const int njt = Nj >> 3;
    for (int jt = 0; jt < njt; ++jt) {
        const int j0 = jt * 8;
        if (jt + 1 < njt) __builtin_prefetch(hj_proj + (j0 + 8) * HIDDEN, 0, 1);
        // stage sender tile
        for (int idx = tid; idx < 8 * HIDDEN; idx += 128)
            sHj[idx >> 6][idx & 63] = (_Float16)hj_proj[(j0 + (idx >> 6)) * HIDDEN + (idx & 63)];
        // geometry per edge (1 edge per thread); edge e = ii*8 + jj
        {
            const int e = tid;
            const int ii = e >> 3, jj = e & 7;
            const int i = i0 + ii, j = j0 + jj;
            float dx = xi[i * 3 + 0] - xj[j * 3 + 0];
            float dy = xi[i * 3 + 1] - xj[j * 3 + 1];
            float dz = xi[i * 3 + 2] - xj[j * 3 + 2];
            float d2 = dx * dx + dy * dy + dz * dz;
            sDiff[e][0] = dx; sDiff[e][1] = dy; sDiff[e][2] = dz;
            sD2[e] = d2;
            float dist = sqrtf(d2 + 1e-12f);
            sAdj[e] = (mi[i] == mj[j] && dist <= cutoff) ? 1.0f : 0.0f;
            sPhi[e] = 0.0f;
        }
        __syncthreads();
        // e = silu(hi + hj + d2*wd2 + eb1)
        for (int idx = tid; idx < 128 * HIDDEN; idx += 128) {
            int e = idx >> 6, f = idx & 63;
            float pre = (float)sHi[e >> 3][f] + (float)sHj[e & 7][f] + sD2[e] * sWd2[f] + sEb1[f];
            sA[e][f] = (_Float16)silu_f(pre);
        }
        __syncthreads();
        // matmul 1: m = silu(e @ W2 + eb2) * adj  -> sM; per-receiver agg reduction
        for (int mt = 0; mt < 8; ++mt) {
            v8f acc = {0.f, 0.f, 0.f, 0.f, 0.f, 0.f, 0.f, 0.f};
            #pragma unroll
            for (int kb = 0; kb < HIDDEN; kb += 32) {
                v16h a, b;
                const int r  = mt * 16 + (lane & 15);
                const int kx = kb + (half << 3);
                const int n  = nb + (lane & 15);
                #pragma unroll
                for (int q = 0; q < 8; ++q) {
                    a[q]     = sA[r][kx + q];
                    a[8 + q] = sA[r][kx + 16 + q];
                    b[q]     = sW2[kx + q][n];
                    b[8 + q] = sW2[kx + 16 + q][n];
                }
                acc = __builtin_amdgcn_wmma_f32_16x16x32_f16(false, a, false, b,
                                                             (short)0, acc, false, false);
            }
            const int n = nb + (lane & 15);
            float s = 0.0f;
            #pragma unroll
            for (int v = 0; v < 8; ++v) {
                int e = mt * 16 + half * 8 + v;          // lane's 8 rows share receiver
                float x = silu_f(acc[v] + sEb2[n]) * sAdj[e];
                sM[e][n] = (_Float16)x;
                s += x;
            }
            sAgg[2 * mt + half][n] += s;                  // unique (ii,n) per lane: no race
        }
        __syncthreads();
        // matmul 2: t = silu(m @ C1 + cb1); phi[e] += t . cw2 (fused dot, LDS atomics)
        for (int mt = 0; mt < 8; ++mt) {
            v8f acc = {0.f, 0.f, 0.f, 0.f, 0.f, 0.f, 0.f, 0.f};
            #pragma unroll
            for (int kb = 0; kb < HIDDEN; kb += 32) {
                v16h a, b;
                const int r  = mt * 16 + (lane & 15);
                const int kx = kb + (half << 3);
                const int n  = nb + (lane & 15);
                #pragma unroll
                for (int q = 0; q < 8; ++q) {
                    a[q]     = sM[r][kx + q];
                    a[8 + q] = sM[r][kx + 16 + q];
                    b[q]     = sC1[kx + q][n];
                    b[8 + q] = sC1[kx + 16 + q][n];
                }
                acc = __builtin_amdgcn_wmma_f32_16x16x32_f16(false, a, false, b,
                                                             (short)0, acc, false, false);
            }
            const int n = nb + (lane & 15);
            #pragma unroll
            for (int v = 0; v < 8; ++v) {
                int e = mt * 16 + half * 8 + v;
                float x = silu_f(acc[v] + sCb1[n]) * sCw2[n];
                atomicAdd(&sPhi[e], x);
            }
        }
        __syncthreads();
        // trans accumulation: diff/(norm+C)*phi*adj  (rcp instead of div)
        {
            const int e = tid;
            float coef = sPhi[e] * sAdj[e] * frcp_f(sqrtf(sD2[e] + 1e-8f) + NORM_CC);
            const int ii = e >> 3;
            atomicAdd(&sTrn[ii][0], sDiff[e][0] * coef);
            atomicAdd(&sTrn[ii][1], sDiff[e][1] * coef);
            atomicAdd(&sTrn[ii][2], sDiff[e][2] * coef);
        }
        __syncthreads();
    }
    // write raw sums (node kernel divides by NORM_F)
    for (int idx = tid; idx < 16 * HIDDEN; idx += 128)
        agg[(i0 + (idx >> 6)) * HIDDEN + (idx & 63)] = sAgg[idx >> 6][idx & 63];
    if (tid < 48) trans[(i0 + tid / 3) * 3 + tid % 3] = sTrn[tid / 3][tid % 3];
}

// node update: h += mlp2(concat(h, agg/100)); x += trans/100   (one wave per node)
__global__ __launch_bounds__(32) void node_update_kernel(
        const float* nw1, const float* nb1, const float* nw2, const float* nb2,
        float* h, float* x, const float* agg, const float* trans) {
    __shared__ float hrow[HIDDEN], arow[HIDDEN], hid[HIDDEN];
    const int i = blockIdx.x;
    const int l = threadIdx.x;
    for (int f = l; f < HIDDEN; f += 32) {
        hrow[f] = h[i * HIDDEN + f];
        arow[f] = agg[i * HIDDEN + f] * (1.0f / NORM_F);
    }
    __syncthreads();
    for (int k = l; k < HIDDEN; k += 32) {
        float a = nb1[k];
        for (int f = 0; f < HIDDEN; ++f) a += hrow[f] * nw1[f * HIDDEN + k];
        for (int f = 0; f < HIDDEN; ++f) a += arow[f] * nw1[(HIDDEN + f) * HIDDEN + k];
        hid[k] = silu_f(a);
    }
    __syncthreads();
    for (int f = l; f < HIDDEN; f += 32) {
        float o = nb2[f];
        for (int k = 0; k < HIDDEN; ++k) o += hid[k] * nw2[k * HIDDEN + f];
        h[i * HIDDEN + f] = hrow[f] + o;
    }
    if (l < 3) x[i * 3 + l] += trans[i * 3 + l] * (1.0f / NORM_F);
}

// output heads + velocity combine + decoder
__global__ void finalize_kernel(
        const float* h_ll, const float* h_cr,
        const float* llwo, const float* llbo,
        const float* crwo, const float* crbo,
        const float* dw1, const float* db1, const float* dw2, const float* db2,
        const float* x_l0, const float* x_ll, const float* x_cr,
        float* out) {
    int i = blockIdx.x * blockDim.x + threadIdx.x;
    if (i >= NLIG) return;
    float ha[17], hb[17];
    for (int f = 0; f < 17; ++f) {
        float a = llbo[f], b = crbo[f];
        for (int k = 0; k < HIDDEN; ++k) {
            a += h_ll[i * HIDDEN + k] * llwo[k * 17 + f];
            b += h_cr[i * HIDDEN + k] * crwo[k * 17 + f];
        }
        ha[f] = a; hb[f] = b;
    }
    float hf[16];
    for (int f = 0; f < 16; ++f) hf[f] = 0.5f * (ha[f] + hb[f]);
    float m1[32];
    for (int k = 0; k < 32; ++k) {
        float a = db1[k];
        for (int f = 0; f < 16; ++f) a += hf[f] * dw1[f * 32 + k];
        m1[k] = silu_f(a);
    }
    for (int d = 0; d < 3; ++d) {
        float va = ntn_f(x_ll[i * 3 + d] - x_l0[i * 3 + d]);
        float vb = ntn_f(x_cr[i * 3 + d] - x_l0[i * 3 + d]);
        out[i * 19 + d] = ntn_f(0.5f * (va + vb));
    }
    for (int f = 0; f < 16; ++f) {
        float a = db2[f];
        for (int k = 0; k < 32; ++k) a += m1[k] * dw2[k * 16 + f];
        out[i * 19 + 3 + f] = ntn_f(a);
    }
}

// ---------------- host orchestration ----------------

extern "C" void kernel_launch(void* const* d_in, const int* in_sizes, int n_in,
                              void* d_out, int out_size, void* d_ws, size_t ws_size,
                              hipStream_t stream) {
    (void)in_sizes; (void)out_size; (void)ws_size;
    if (n_in < 72) return;

    const float* xh_lig = (const float*)d_in[0];
    const float* xh_ctx = (const float*)d_in[1];
    const float* tvec   = (const float*)d_in[2];
    const float* noise  = (const float*)d_in[3];
    // params pytree (dict keys sorted): cr, dec, enc_a, enc_p, ll
    const float* cr_bl  = (const float*)d_in[4];
    const float* cr_bo  = (const float*)d_in[5];
    const float* cr_bp  = (const float*)d_in[6];
    // cr layers at base 7 and 18 (cb1,cw1,cw2,eb1,eb2,ew1,ew2,nb1,nb2,nw1,nw2)
    const float* cr_wl  = (const float*)d_in[29];
    const float* cr_wo  = (const float*)d_in[30];
    const float* cr_wp  = (const float*)d_in[31];
    const float* dec_w1 = (const float*)d_in[32];
    const float* dec_b1 = (const float*)d_in[33];
    const float* dec_w2 = (const float*)d_in[34];
    const float* dec_b2 = (const float*)d_in[35];
    const float* ea_w1  = (const float*)d_in[36];
    const float* ea_b1  = (const float*)d_in[37];
    const float* ea_w2  = (const float*)d_in[38];
    const float* ea_b2  = (const float*)d_in[39];
    const float* ep_w1  = (const float*)d_in[40];
    const float* ep_b1  = (const float*)d_in[41];
    const float* ep_w2  = (const float*)d_in[42];
    const float* ep_b2  = (const float*)d_in[43];
    const float* ll_bi  = (const float*)d_in[44];
    const float* ll_bo  = (const float*)d_in[45];
    // ll layers at base 46 and 57
    const float* ll_wi  = (const float*)d_in[68];
    const float* ll_wo  = (const float*)d_in[69];
    const int* mask_lig = (const int*)d_in[70];
    const int* mask_ctx = (const int*)d_in[71];

    float* out = (float*)d_out;
    float* ws  = (float*)d_ws;

    // workspace layout (float offsets)
    float* x_l0    = ws;                  // 512*3
    float* x_ll    = x_l0 + 1536;         // 512*3
    float* x_cr    = x_ll + 1536;         // 512*3
    float* h_lt    = x_cr + 1536;         // 512*17
    float* h_p_emb = h_lt + 512 * 17;     // 1024*16
    float* hp_proj = h_p_emb + 16384;     // 1024*64
    float* h_ll    = hp_proj + 65536;     // 512*64
    float* h_cr    = h_ll + 32768;        // 512*64
    float* hi_prj  = h_cr + 32768;        // 512*64
    float* hj_prj  = hi_prj + 32768;      // 1024*64
    float* agg     = hj_prj + 65536;      // 512*64
    float* trn     = agg + 32768;         // 512*3
    float* x_p     = trn + 1536;          // 1024*3

    auto lp = [&](int base, int off) { return (const float*)d_in[base + off]; };
    // layer leaf order: 0 cb1, 1 cw1, 2 cw2, 3 eb1, 4 eb2, 5 ew1, 6 ew2, 7 nb1, 8 nb2, 9 nw1, 10 nw2

    // zero the context half of the output
    zero_kernel<<<(NPOC * 35 + 255) / 256, 256, 0, stream>>>(out + NLIG * 19, NPOC * 35);

    enc_lig_kernel<<<2, 256, 0, stream>>>(xh_lig, tvec, mask_lig, ea_w1, ea_b1, ea_w2, ea_b2, h_lt);
    enc_poc_kernel<<<4, 256, 0, stream>>>(xh_ctx, ep_w1, ep_b1, ep_w2, ep_b2, cr_wp, cr_bp,
                                          h_p_emb, hp_proj, x_p);
    lig_init_kernel<<<2, 256, 0, stream>>>(xh_lig, noise, h_lt, ll_wi, ll_bi, cr_wl, cr_bl,
                                           x_l0, x_ll, x_cr, h_ll, h_cr);

    auto run_layer = [&](int base, float* h_rcv, const float* h_snd, int Nj,
                         float* x_rcv, const float* x_snd, const int* mj, float cutoff) {
        const float* ew1 = lp(base, 5);
        proj_kernel<<<(NLIG * HIDDEN + 255) / 256, 256, 0, stream>>>(NLIG, h_rcv, ew1, hi_prj);
        proj_kernel<<<(Nj * HIDDEN + 255) / 256, 256, 0, stream>>>(Nj, h_snd, ew1 + 64 * 64, hj_prj);
        egcl_edge_kernel<<<NLIG / 16, 128, 0, stream>>>(
            Nj, hi_prj, hj_prj, x_rcv, x_snd, mask_lig, mj, cutoff,
            ew1, lp(base, 3), lp(base, 6), lp(base, 4),
            lp(base, 1), lp(base, 0), lp(base, 2), agg, trn);
        node_update_kernel<<<NLIG, 32, 0, stream>>>(
            lp(base, 9), lp(base, 7), lp(base, 10), lp(base, 8), h_rcv, x_rcv, agg, trn);
    };

    // ligand-ligand branch (2 layers), cutoff 2.0
    run_layer(46, h_ll, h_ll, NLIG, x_ll, x_ll, mask_lig, 2.0f);
    run_layer(57, h_ll, h_ll, NLIG, x_ll, x_ll, mask_lig, 2.0f);
    // cross branch (2 layers), fixed pocket senders, cutoff 2.0
    run_layer(7,  h_cr, hp_proj, NPOC, x_cr, x_p, mask_ctx, 2.0f);
    run_layer(18, h_cr, hp_proj, NPOC, x_cr, x_p, mask_ctx, 2.0f);

    finalize_kernel<<<2, 256, 0, stream>>>(h_ll, h_cr, ll_wo, ll_bo, cr_wo, cr_bo,
                                           dec_w1, dec_b1, dec_w2, dec_b2,
                                           x_l0, x_ll, x_cr, out);
}
